// SwitchTransformerEncoderLayer_80814104642411
// MI455X (gfx1250) — matrix operand order repaired and checked by reading, compiled, not run on hardware
//
#include <hip/hip_runtime.h>
#include <math.h>

// ---------------- problem constants ----------------
constexpr int S_ = 1024, B_ = 4, D_ = 1024, H_ = 4, E_ = 32, DH_ = 4096;
constexpr int CAP_ = 512, T_ = 4096, HD_ = 256;   // T = S*B, HD = D/H

// ---------------- CDNA5 WMMA types ----------------
typedef __attribute__((ext_vector_type(16))) __bf16 v16bf;
typedef __attribute__((ext_vector_type(8)))  __bf16 v8bf;
typedef __attribute__((ext_vector_type(8)))  float  v8f;
typedef unsigned short u16;
union V16 { v16bf v; v8bf h[2]; };

__device__ __forceinline__ u16 f2bf(float f) {
  unsigned u = __float_as_uint(f);
  unsigned r = (u + 0x7FFFu + ((u >> 16) & 1u)) >> 16;
  return (u16)r;
}

// ---- CDNA5 async LDS copy (ASYNCcnt-tracked DMA, ISA 15.18.3 op 98) ----
__device__ __forceinline__ void async_b128(const u16* g, u16* l) {
  unsigned loff = (unsigned)(unsigned long long)l;  // flat LDS aperture: low 32 = LDS byte addr
  asm volatile("global_load_async_to_lds_b128 %0, %1, off"
               :: "v"(loff), "v"((unsigned long long)g) : "memory");
}
__device__ __forceinline__ void wait_async0() {
  asm volatile("s_wait_asynccnt 0" ::: "memory");
}

// ---------------- epilogue kinds ----------------
constexpr int EP_SCALE = 0;        // C(f32) = acc * scale
constexpr int EP_BIAS_RES = 1;     // C(f32) = acc + bias + R
constexpr int EP_BIAS_SCATTER = 2; // moe_out[slot_token[row]] = acc + bias
constexpr int EP_BIAS_BF16 = 3;    // C(bf16) = acc + bias
constexpr int EP_STORE_BF16 = 4;   // C(bf16) = acc
constexpr int EP_GELU_BF16 = 5;    // C(bf16) = gelu(acc + bias)

struct GemmP {
  const u16*   A;      // bf16, row-major [M,K], k contiguous, stride ldA
  const u16*   Bm;     // bf16, element (n,k) at n*ldBn + k, k contiguous
  const float* bias;   // may be null
  const float* R;      // residual (f32), may be null
  const int*   aux;    // slot_token for scatter
  void*        C;
  int M, N, K;
  long ldA, ldBn, ldC;
  int  shA, mkA, shB, mkB, shC, mkC;   // z decomposition: hi = z>>sh, lo = z&mk
  long zA1, zA2, zB1, zB2, zC1, zC2, zBias;
  float scale;
};

// Tiled bf16 WMMA GEMM, async double-buffered LDS staging.
// Block: 256 threads = 8 waves; tile 128x128x32; wave tile 64x32 (4x2 WMMA 16x16x32).
template <int EP>
__global__ __launch_bounds__(256) void gemm_k(GemmP p) {
  constexpr int BM = 128, BN = 128, BK = 32, BKP = 40; // 80B LDS row: 16B aligned, bank-spread
  __shared__ __align__(16) u16 As[2][BM][BKP];
  __shared__ __align__(16) u16 Bs[2][BN][BKP];

  const int tid = threadIdx.x;
  const int z = blockIdx.z;
  const long zA = (long)(z >> p.shA) * p.zA1 + (long)(z & p.mkA) * p.zA2;
  const long zB = (long)(z >> p.shB) * p.zB1 + (long)(z & p.mkB) * p.zB2;
  const long zC = (long)(z >> p.shC) * p.zC1 + (long)(z & p.mkC) * p.zC2;
  const long m0 = (long)blockIdx.y * BM;
  const long n0 = (long)blockIdx.x * BN;

  const int wave = tid >> 5, lane = tid & 31;
  const int wm = wave >> 2, wn = wave & 3;       // 2 (M) x 4 (N) wave grid
  const int half = lane >> 4, mr = lane & 15;

  const u16* aBase = p.A + zA;
  const u16* bBase = p.Bm + zB;

  // stage one 128x32 bf16 tile pair via async DMA: 4 b128 chunks/row, 2 rows/thread
  auto stage = [&](int buf, int k0) {
    int idx = tid;
#pragma unroll
    for (int it = 0; it < 2; ++it, idx += 256) {
      int r = idx >> 2, c = (idx & 3) * 8;
      async_b128(aBase + (m0 + r) * p.ldA + k0 + c, &As[buf][r][c]);
    }
    idx = tid;
#pragma unroll
    for (int it = 0; it < 2; ++it, idx += 256) {
      int r = idx >> 2, c = (idx & 3) * 8;
      async_b128(bBase + (n0 + r) * p.ldBn + k0 + c, &Bs[buf][r][c]);
    }
  };

  v8f acc[4][2] = {};
  const int nk = p.K / BK;
  stage(0, 0);

  for (int kt = 0; kt < nk; ++kt) {
    const int cur = kt & 1;
    wait_async0();       // my async writes for tile kt are in LDS
    __syncthreads();     // everyone's are; everyone done reading tile kt-1's buffer
    if (kt + 1 < nk) stage(cur ^ 1, (kt + 1) * BK);  // DMA next tile under WMMA

    V16 a[4], b[2];
#pragma unroll
    for (int mt = 0; mt < 4; ++mt) {
      int r = wm * 64 + mt * 16 + mr;
      a[mt].h[0] = *(const v8bf*)&As[cur][r][half * 8];
      a[mt].h[1] = *(const v8bf*)&As[cur][r][16 + half * 8];
    }
#pragma unroll
    for (int nt = 0; nt < 2; ++nt) {
      int c = wn * 32 + nt * 16 + mr;
      b[nt].h[0] = *(const v8bf*)&Bs[cur][c][half * 8];
      b[nt].h[1] = *(const v8bf*)&Bs[cur][c][16 + half * 8];
    }
#pragma unroll
    for (int mt = 0; mt < 4; ++mt)
#pragma unroll
      for (int nt = 0; nt < 2; ++nt)
        acc[mt][nt] = __builtin_amdgcn_wmma_f32_16x16x32_bf16(
            false, a[mt].v, false, b[nt].v, (short)0, acc[mt][nt], false, false);
  }

  // ---- epilogue ----
#pragma unroll
  for (int mt = 0; mt < 4; ++mt)
#pragma unroll
    for (int nt = 0; nt < 2; ++nt)
#pragma unroll
      for (int r = 0; r < 8; ++r) {
        long row = m0 + wm * 64 + mt * 16 + half * 8 + r;
        long col = n0 + wn * 32 + nt * 16 + mr;
        float v = acc[mt][nt][r];
        if constexpr (EP == EP_SCALE) v *= p.scale;
        if constexpr (EP == EP_BIAS_RES || EP == EP_BIAS_SCATTER ||
                      EP == EP_BIAS_BF16 || EP == EP_GELU_BF16)
          v += p.bias[(long)z * p.zBias + col];
        if constexpr (EP == EP_BIAS_RES) v += p.R[row * p.ldC + col];
        if constexpr (EP == EP_GELU_BF16) {
          v = 0.5f * v * (1.0f + erff(v * 0.70710678118f));  // exact gelu
          ((u16*)p.C)[zC + row * p.ldC + col] = f2bf(v);
        } else if constexpr (EP == EP_BIAS_BF16 || EP == EP_STORE_BF16) {
          ((u16*)p.C)[zC + row * p.ldC + col] = f2bf(v);
        } else if constexpr (EP == EP_BIAS_SCATTER) {
          int t = p.aux[(long)z * CAP_ + (int)row];
          if (t >= 0) ((float*)p.C)[(long)t * p.ldC + col] = v;
        } else {
          ((float*)p.C)[zC + row * p.ldC + col] = v;
        }
      }
}

// ---------------- f32 -> bf16 bulk convert ----------------
__global__ __launch_bounds__(256) void cvt_k(const float* src, u16* dst, long n) {
  long i = ((long)blockIdx.x * 256 + threadIdx.x) * 4;
  if (i >= n) return;
  float4 v = *(const float4*)(src + i);
  unsigned lo = (unsigned)f2bf(v.x) | ((unsigned)f2bf(v.y) << 16);
  unsigned hi = (unsigned)f2bf(v.z) | ((unsigned)f2bf(v.w) << 16);
  uint2 o; o.x = lo; o.y = hi;
  *(uint2*)(dst + i) = o;
}

// ---------------- V transpose: qkv(bf16) -> vT[b,h,dv,s] (bf16) ----------------
__global__ __launch_bounds__(256) void vtrans_k(const u16* qkvb, u16* vT) {
  __shared__ u16 t[32][33];
  int z = blockIdx.z, b = z >> 2, h = z & 3;
  int s0 = blockIdx.x * 32, d0 = blockIdx.y * 32;
  int tr = threadIdx.x >> 5, tc = threadIdx.x & 31;
#pragma unroll
  for (int j = 0; j < 4; ++j) {
    int sl = tr + j * 8;
    t[sl][tc] = qkvb[((long)(s0 + sl) * B_ + b) * (3 * D_) + 2 * D_ + h * HD_ + d0 + tc];
  }
  __syncthreads();
#pragma unroll
  for (int j = 0; j < 4; ++j) {
    int dl = tr + j * 8;
    vT[((long)z * HD_ + d0 + dl) * S_ + s0 + tc] = t[tc][dl];
  }
}

// ---------------- row softmax (wave per 1024-wide row), bf16 out ----------------
__global__ __launch_bounds__(256) void softmax_k(const float* X, u16* P, int rows) {
  int wave = threadIdx.x >> 5, lane = threadIdx.x & 31;
  int row = blockIdx.x * 8 + wave;
  if (row >= rows) return;
  const float* x = X + (long)row * 1024;
  float vals[32], mx = -3.4e38f;
#pragma unroll
  for (int i = 0; i < 32; ++i) { vals[i] = x[lane + i * 32]; mx = fmaxf(mx, vals[i]); }
  for (int m = 16; m > 0; m >>= 1) mx = fmaxf(mx, __shfl_xor(mx, m, 32));
  float s = 0.f;
#pragma unroll
  for (int i = 0; i < 32; ++i) { vals[i] = __expf(vals[i] - mx); s += vals[i]; }
  for (int m = 16; m > 0; m >>= 1) s += __shfl_xor(s, m, 32);
  float inv = 1.f / s;
#pragma unroll
  for (int i = 0; i < 32; ++i) P[(long)row * 1024 + lane + i * 32] = f2bf(vals[i] * inv);
}

// ---------------- LayerNorm (wave per row, optional residual) ----------------
__global__ __launch_bounds__(256) void ln_k(const float* X, const float* Radd,
                                            const float* g, const float* b,
                                            float* out, int rows) {
  int wave = threadIdx.x >> 5, lane = threadIdx.x & 31;
  int row = blockIdx.x * 8 + wave;
  if (row >= rows) return;
  const float* x = X + (long)row * 1024;
  float vals[32], s = 0.f, sq = 0.f;
#pragma unroll
  for (int i = 0; i < 32; ++i) {
    int c = lane + i * 32;
    float v = x[c] + (Radd ? Radd[(long)row * 1024 + c] : 0.f);
    vals[i] = v; s += v; sq += v * v;
  }
  for (int m = 16; m > 0; m >>= 1) { s += __shfl_xor(s, m, 32); sq += __shfl_xor(sq, m, 32); }
  float mu = s * (1.f / 1024.f);
  float var = sq * (1.f / 1024.f) - mu * mu;
  float inv = rsqrtf(var + 1e-5f);
#pragma unroll
  for (int i = 0; i < 32; ++i) {
    int c = lane + i * 32;
    out[(long)row * 1024 + c] = (vals[i] - mu) * inv * g[c] + b[c];
  }
}

// ---------------- MoE state init ----------------
__global__ __launch_bounds__(256) void init_k(int* cnt, int* slot, float* moe_out, long n) {
  long i = (long)blockIdx.x * 256 + threadIdx.x;
  if (i < E_) cnt[i] = 0;
  if (i < (long)E_ * CAP_) slot[i] = -1;
  if (i < n) moe_out[i] = 0.f;
}

// ---------------- gate: top-1 argmax + capacity dispatch (wave per token) ----------------
__global__ __launch_bounds__(256) void gate_k(const float* x1, const float* gw,
                                              const float* gb, int* cnt, int* slot,
                                              u16* buf) {
  int wave = threadIdx.x >> 5, lane = threadIdx.x & 31;
  int t = blockIdx.x * 8 + wave;
  if (t >= T_) return;
  const float4* xr = (const float4*)(x1 + (long)t * 1024);
  const float4* wr = (const float4*)(gw + (long)lane * 1024);  // lane == expert (E=32)
  float acc = gb[lane];
  for (int i = 0; i < 256; ++i) {
    float4 a = xr[i], w = wr[i];
    acc += a.x * w.x + a.y * w.y + a.z * w.z + a.w * w.w;
  }
  float v = acc; int e = lane;
  for (int m = 16; m > 0; m >>= 1) {
    float ov = __shfl_xor(v, m, 32);
    int   oe = __shfl_xor(e, m, 32);
    if (ov > v || (ov == v && oe < e)) { v = ov; e = oe; }  // first-max like argmax
  }
  int c = 0;
  if (lane == 0) c = atomicAdd(&cnt[e], 1);
  c = __shfl(c, 0, 32);
  if (c < CAP_) {
    if (lane == 0) slot[e * CAP_ + c] = t;
    u16* dst = buf + ((long)e * CAP_ + c) * 1024;
    const float* src = x1 + (long)t * 1024;
#pragma unroll
    for (int i = 0; i < 32; ++i) dst[lane + i * 32] = f2bf(src[lane + i * 32]);
  }
}

// ---------------- host-side orchestration ----------------
extern "C" void kernel_launch(void* const* d_in, const int* in_sizes, int n_in,
                              void* d_out, int out_size, void* d_ws, size_t ws_size,
                              hipStream_t stream) {
  const float* x      = (const float*)d_in[0];
  const float* in_w   = (const float*)d_in[1];
  const float* in_b   = (const float*)d_in[2];
  const float* out_w  = (const float*)d_in[3];
  const float* out_b  = (const float*)d_in[4];
  const float* gate_w = (const float*)d_in[5];
  const float* gate_b = (const float*)d_in[6];
  const float* w1     = (const float*)d_in[7];
  const float* b1     = (const float*)d_in[8];
  const float* w2     = (const float*)d_in[9];
  const float* b2     = (const float*)d_in[10];
  const float* ln1_g  = (const float*)d_in[11];
  const float* ln1_b  = (const float*)d_in[12];
  const float* ln2_g  = (const float*)d_in[13];
  const float* ln2_b  = (const float*)d_in[14];
  float* out = (float*)d_out;

  // workspace layout (MB offsets); phase-B regions reuse dead phase-A regions
  const size_t MB = 1024ull * 1024ull;
  char* ws = (char*)d_ws;
  u16*   w1b    = (u16*)  (ws + 0);              // 256MB, whole launch
  u16*   w2b    = (u16*)  (ws + 256 * MB);       // 256MB, whole launch
  u16*   inwb   = (u16*)  (ws + 512 * MB);       // 6MB
  u16*   outwb  = (u16*)  (ws + 520 * MB);       // 2MB
  u16*   xb     = (u16*)  (ws + 524 * MB);       // 8MB
  u16*   qkvb   = (u16*)  (ws + 532 * MB);       // 24MB
  float* scores = (float*)(ws + 556 * MB);       // 64MB  (phase A)
  u16*   buf    = (u16*)  (ws + 556 * MB);       // 32MB  (phase B, over scores)
  u16*   hbuf   = (u16*)  (ws + 588 * MB);       // 128MB (phase B, over probs/vT/ob/x1pre)
  u16*   probs  = (u16*)  (ws + 620 * MB);       // 32MB  (phase A)
  u16*   vT     = (u16*)  (ws + 652 * MB);       // 8MB  (phase A)
  u16*   ob     = (u16*)  (ws + 684 * MB);       // 8MB   (phase A)
  float* x1pre  = (float*)(ws + 692 * MB);       // 16MB  (phase A)
  float* x1     = (float*)(ws + 716 * MB);       // 16MB
  float* moe_o  = (float*)(ws + 732 * MB);       // 16MB
  int*   cnt    = (int*)  (ws + 748 * MB);       // [E]
  int*   slot   = (int*)  (ws + 748 * MB + 256); // [E,CAP]

  const long ld3D = 3L * D_;
  const long ldQKV = (long)B_ * 3 * D_;

  // 0) one-time f32->bf16 conversions (weights + x)
  cvt_k<<<dim3(3072),   256, 0, stream>>>(in_w,  inwb,  3L * D_ * D_);
  cvt_k<<<dim3(1024),   256, 0, stream>>>(out_w, outwb, (long)D_ * D_);
  cvt_k<<<dim3(131072), 256, 0, stream>>>(w1,    w1b,   (long)E_ * DH_ * D_);
  cvt_k<<<dim3(131072), 256, 0, stream>>>(w2,    w2b,   (long)E_ * D_ * DH_);
  cvt_k<<<dim3(4096),   256, 0, stream>>>(x,     xb,    (long)T_ * D_);

  GemmP p;

  // 1) qkv(bf16) = x @ in_w^T + in_b
  p = {}; p.A = xb; p.Bm = inwb; p.bias = in_b; p.C = qkvb;
  p.M = T_; p.N = 3 * D_; p.K = D_;
  p.ldA = D_; p.ldBn = D_; p.ldC = 3 * D_; p.zA1 = p.zB1 = p.zC1 = 0;
  gemm_k<EP_BIAS_BF16><<<dim3(24, 32, 1), 256, 0, stream>>>(p);

  // 2) scores(f32) = (Q @ K^T) / 16    per z=(b*H+h)
  p = {}; p.A = qkvb; p.Bm = qkvb + D_; p.C = scores;
  p.M = S_; p.N = S_; p.K = HD_;
  p.ldA = ldQKV; p.ldBn = ldQKV; p.ldC = S_;
  p.shA = 2; p.mkA = 3; p.zA1 = ld3D; p.zA2 = HD_;
  p.shB = 2; p.mkB = 3; p.zB1 = ld3D; p.zB2 = HD_;
  p.zC1 = (long)S_ * S_;
  p.scale = 1.0f / 16.0f;
  gemm_k<EP_SCALE><<<dim3(8, 8, B_ * H_), 256, 0, stream>>>(p);

  // 3) probs(bf16) = softmax(scores)
  softmax_k<<<dim3((B_ * H_ * S_) / 8), 256, 0, stream>>>(scores, probs, B_ * H_ * S_);

  // 4) vT(bf16): [b,h,dv,s]
  vtrans_k<<<dim3(32, 8, B_ * H_), 256, 0, stream>>>(qkvb, vT);

  // 5) ob(bf16) = P @ V   per (b,h), written as flat [T,D]
  p = {}; p.A = probs; p.Bm = vT; p.C = ob;
  p.M = S_; p.N = HD_; p.K = S_;
  p.ldA = S_; p.ldBn = S_; p.ldC = (long)B_ * D_;
  p.zA1 = (long)S_ * S_;
  p.zB1 = (long)HD_ * S_;
  p.shC = 2; p.mkC = 3; p.zC1 = D_; p.zC2 = HD_;
  gemm_k<EP_STORE_BF16><<<dim3(2, 8, B_ * H_), 256, 0, stream>>>(p);

  // 6) x1pre(f32) = ob @ out_w^T + out_b + x
  p = {}; p.A = ob; p.Bm = outwb; p.bias = out_b; p.R = x; p.C = x1pre;
  p.M = T_; p.N = D_; p.K = D_;
  p.ldA = D_; p.ldBn = D_; p.ldC = D_;
  gemm_k<EP_BIAS_RES><<<dim3(8, 32, 1), 256, 0, stream>>>(p);

  // 7) x1 = LN1(x1pre)
  ln_k<<<dim3(T_ / 8), 256, 0, stream>>>(x1pre, nullptr, ln1_g, ln1_b, x1, T_);

  // 8) init MoE state
  init_k<<<dim3(((long)T_ * D_) / 256), 256, 0, stream>>>(cnt, slot, moe_o, (long)T_ * D_);

  // 9) gate + dispatch (buf bf16)
  gate_k<<<dim3(T_ / 8), 256, 0, stream>>>(x1, gate_w, gate_b, cnt, slot, buf);

  // 10) h(bf16) = gelu(buf @ w1^T + b1)   per expert
  p = {}; p.A = buf; p.Bm = w1b; p.bias = b1; p.C = hbuf;
  p.M = CAP_; p.N = DH_; p.K = D_;
  p.ldA = D_; p.ldBn = D_; p.ldC = DH_;
  p.zA1 = (long)CAP_ * D_;
  p.zB1 = (long)DH_ * D_;
  p.zC1 = (long)CAP_ * DH_;
  p.zBias = DH_;
  gemm_k<EP_GELU_BF16><<<dim3(32, 4, E_), 256, 0, stream>>>(p);

  // 11) moe_o[token] = h @ w2^T + b2  (scatter via slot_token)
  p = {}; p.A = hbuf; p.Bm = w2b; p.bias = b2; p.aux = slot; p.C = moe_o;
  p.M = CAP_; p.N = D_; p.K = DH_;
  p.ldA = DH_; p.ldBn = DH_; p.ldC = D_;
  p.zA1 = (long)CAP_ * DH_;
  p.zB1 = (long)D_ * DH_;
  p.zBias = D_;
  gemm_k<EP_BIAS_SCATTER><<<dim3(8, 4, E_), 256, 0, stream>>>(p);

  // 12) out = LN2(x1 + moe_o)
  ln_k<<<dim3(T_ / 8), 256, 0, stream>>>(x1, moe_o, ln2_g, ln2_b, out, T_);
}
